// LSTM_cell_39015482916921
// MI455X (gfx1250) — matrix operand order repaired and checked by reading, not compiled, not run on hardware
//
#include <hip/hip_runtime.h>
#include <hip/hip_bf16.h>

// LSTM cell, fused 4-gate GEMM via bf16 WMMA (gfx1250 / wave32).
//   gates = [i|h_prev] @ [Wf;Wi;Wc;Wo]^T + b      (M=16384, N=4*512, K=1024)
//   c_new = sig(f)*c_prev + sig(i)*tanh(c~);  h_new = sig(o)*tanh(c_new)

#define B_SZ   16384
#define IN_SZ  512
#define H_SZ   512
#define K_SZ   (IN_SZ + H_SZ)   // 1024
#define KC     32                // K per WMMA step (bf16 K=32)
#define NKC    (K_SZ / KC)       // 32
#define MT     128               // batch rows per workgroup
#define HT     32                // h columns per workgroup
#define SA     40                // padded LDS row stride (bf16 elems, 80B)

typedef __attribute__((ext_vector_type(16))) __bf16 v16bf;
typedef __attribute__((ext_vector_type(8)))  __bf16 v8bf;
typedef __attribute__((ext_vector_type(8)))  float  v8f;

union V16U { v16bf v; v8bf h[2]; };

__device__ __forceinline__ float sigmoid_f(float x) {
  return 1.0f / (1.0f + __expf(-x));
}
__device__ __forceinline__ float tanh_f(float x) {
  return 1.0f - 2.0f / (__expf(2.0f * x) + 1.0f);
}

__global__ __launch_bounds__(256)
void lstm_cell_wmma(const float* __restrict__ xin,
                    const float* __restrict__ hprev,
                    const float* __restrict__ cprev,
                    const float* __restrict__ Wf, const float* __restrict__ bfp,
                    const float* __restrict__ Wi, const float* __restrict__ bip,
                    const float* __restrict__ Wc, const float* __restrict__ bcp,
                    const float* __restrict__ Wo, const float* __restrict__ bop,
                    float* __restrict__ out)
{
  // A tile: 128 rows x 32 k ; W tile: (4 gates * 32 cols) x 32 k ; double buffered
  __shared__ __align__(16) __bf16 sA[2][MT * SA];
  __shared__ __align__(16) __bf16 sW[2][MT * SA];

  const int tid    = threadIdx.x;
  const int lane   = tid & 31;
  const int wave   = tid >> 5;       // 0..7
  const int wave_m = wave >> 1;      // 0..3 : 32-row group
  const int wave_h = wave & 1;       // 0..1 : 16-col group
  const int l16    = lane & 15;
  const int hi16   = lane >> 4;      // 0 or 1 (half-wave)

  const int tile_h = blockIdx.x;     // 0..15
  const int tile_m = blockIdx.y;     // 0..127

  // ---- cooperative-load geometry: 2 threads per row, 16 floats each ----
  const int ldrow = tid >> 1;            // 0..127
  const int ldoff = (tid & 1) * 16;      // k offset within chunk: 0 or 16

  const int a_m = tile_m * MT + ldrow;   // global batch row for A
  const int w_g = ldrow >> 5;            // gate id for W rows
  const int w_c = ldrow & 31;            // col within tile
  const int w_n = tile_h * HT + w_c;     // output neuron index
  const float* Wsel = (w_g == 0) ? Wf : (w_g == 1) ? Wi : (w_g == 2) ? Wc : Wo;
  const float* wrow = Wsel + (size_t)w_n * K_SZ;

  // ---- accumulators: [gate][row-group], each 16x16 f32 = 8 VGPRs ----
  v8f acc[4][2];
  #pragma unroll
  for (int g = 0; g < 4; ++g)
    #pragma unroll
    for (int r = 0; r < 2; ++r)
      acc[g][r] = v8f{};

  float4 ra[4], rw[4];  // staging registers for next chunk

  auto gload = [&](int kc) {
    const int kg = kc * KC + ldoff;   // 0..1023, 64B aligned
    const float* asrc = (kg < IN_SZ)
        ? (xin   + (size_t)a_m * IN_SZ + kg)
        : (hprev + (size_t)a_m * H_SZ  + (kg - IN_SZ));
    const float4* ap = (const float4*)asrc;
    ra[0] = ap[0]; ra[1] = ap[1]; ra[2] = ap[2]; ra[3] = ap[3];
    const float4* wp = (const float4*)(wrow + kg);
    rw[0] = wp[0]; rw[1] = wp[1]; rw[2] = wp[2]; rw[3] = wp[3];
  };

  auto lstore = [&](int buf) {
    v8bf p0, p1, q0, q1;
    p0[0]=(__bf16)ra[0].x; p0[1]=(__bf16)ra[0].y; p0[2]=(__bf16)ra[0].z; p0[3]=(__bf16)ra[0].w;
    p0[4]=(__bf16)ra[1].x; p0[5]=(__bf16)ra[1].y; p0[6]=(__bf16)ra[1].z; p0[7]=(__bf16)ra[1].w;
    p1[0]=(__bf16)ra[2].x; p1[1]=(__bf16)ra[2].y; p1[2]=(__bf16)ra[2].z; p1[3]=(__bf16)ra[2].w;
    p1[4]=(__bf16)ra[3].x; p1[5]=(__bf16)ra[3].y; p1[6]=(__bf16)ra[3].z; p1[7]=(__bf16)ra[3].w;
    q0[0]=(__bf16)rw[0].x; q0[1]=(__bf16)rw[0].y; q0[2]=(__bf16)rw[0].z; q0[3]=(__bf16)rw[0].w;
    q0[4]=(__bf16)rw[1].x; q0[5]=(__bf16)rw[1].y; q0[6]=(__bf16)rw[1].z; q0[7]=(__bf16)rw[1].w;
    q1[0]=(__bf16)rw[2].x; q1[1]=(__bf16)rw[2].y; q1[2]=(__bf16)rw[2].z; q1[3]=(__bf16)rw[2].w;
    q1[4]=(__bf16)rw[3].x; q1[5]=(__bf16)rw[3].y; q1[6]=(__bf16)rw[3].z; q1[7]=(__bf16)rw[3].w;
    __bf16* da = &sA[buf][ldrow * SA + ldoff];
    *(v8bf*)da       = p0;
    *(v8bf*)(da + 8) = p1;
    __bf16* dw = &sW[buf][ldrow * SA + ldoff];
    *(v8bf*)dw       = q0;
    *(v8bf*)(dw + 8) = q1;
  };

  auto compute = [&](int buf) {
    // B fragments: lane = column n (l16); K chunks at (lane<16?0:8) and +16
    V16U bfr[4];
    #pragma unroll
    for (int g = 0; g < 4; ++g) {
      const __bf16* pb = &sW[buf][(g * 32 + wave_h * 16 + l16) * SA + hi16 * 8];
      bfr[g].h[0] = *(const v8bf*)pb;
      bfr[g].h[1] = *(const v8bf*)(pb + 16);
    }
    #pragma unroll
    for (int r = 0; r < 2; ++r) {
      // A fragment: lane = row m (l16); K chunks at (lane<16?0:8) and +16
      const __bf16* pa = &sA[buf][(wave_m * 32 + r * 16 + l16) * SA + hi16 * 8];
      V16U af;
      af.h[0] = *(const v8bf*)pa;
      af.h[1] = *(const v8bf*)(pa + 16);
      #pragma unroll
      for (int g = 0; g < 4; ++g)
        acc[g][r] = __builtin_amdgcn_wmma_f32_16x16x32_bf16(
            false, af.v, false, bfr[g].v, (short)0, acc[g][r], false, false);
    }
  };

  // ---- pipelined K loop (double-buffered LDS) ----
  gload(0);
  lstore(0);
  __syncthreads();
  #pragma unroll 1
  for (int kc = 0; kc < NKC; ++kc) {
    if (kc + 1 < NKC) gload(kc + 1);       // overlap next chunk's VMEM with WMMA
    compute(kc & 1);
    if (kc + 1 < NKC) lstore((kc + 1) & 1);
    __syncthreads();
  }

  // ---- epilogue: bias + activations + cell update, fused ----
  const int n = tile_h * HT + wave_h * 16 + l16;   // h index 0..511
  const float bfv = bfp[n], biv = bip[n], bcv = bcp[n], bov = bop[n];
  const int m0     = tile_m * MT + wave_m * 32;
  const int rowoff = hi16 * 8;                     // C/D layout: lanes>=16 hold M+8
  #pragma unroll
  for (int r = 0; r < 2; ++r) {
    #pragma unroll
    for (int e = 0; e < 8; ++e) {
      const int m = m0 + r * 16 + rowoff + e;
      const float gf = sigmoid_f(acc[0][r][e] + bfv);
      const float gi = sigmoid_f(acc[1][r][e] + biv);
      const float gc = tanh_f   (acc[2][r][e] + bcv);
      const float go = sigmoid_f(acc[3][r][e] + bov);
      const float cp = cprev[(size_t)m * H_SZ + n];
      const float cn = gf * cp + gi * gc;
      const float hn = go * tanh_f(cn);
      out[(size_t)m * H_SZ + n]                          = hn;  // h_new
      out[(size_t)B_SZ * H_SZ + (size_t)m * H_SZ + n]    = cn;  // c_new
    }
  }
}

extern "C" void kernel_launch(void* const* d_in, const int* in_sizes, int n_in,
                              void* d_out, int out_size, void* d_ws, size_t ws_size,
                              hipStream_t stream) {
  const float* xi    = (const float*)d_in[0];
  const float* hprev = (const float*)d_in[1];
  const float* cprev = (const float*)d_in[2];
  const float* Wf    = (const float*)d_in[3];
  const float* bf    = (const float*)d_in[4];
  const float* Wi    = (const float*)d_in[5];
  const float* bi    = (const float*)d_in[6];
  const float* Wc    = (const float*)d_in[7];
  const float* bc    = (const float*)d_in[8];
  const float* Wo    = (const float*)d_in[9];
  const float* bo    = (const float*)d_in[10];
  float* out = (float*)d_out;

  dim3 grid(H_SZ / HT, B_SZ / MT);   // (16, 128)
  lstm_cell_wmma<<<grid, dim3(256), 0, stream>>>(
      xi, hprev, cprev, Wf, bf, Wi, bi, Wc, bc, Wo, bo, out);
}
